// TransmorpherLayer2d_11931419148698
// MI455X (gfx1250) — matrix operand order repaired and checked by reading, not compiled, not run on hardware
//
#include <hip/hip_runtime.h>
#include <hip/hip_bf16.h>
#include <cstdint>
#include <cstddef>

typedef unsigned short u16b;
typedef __attribute__((ext_vector_type(16))) __bf16 v16bf;
typedef __attribute__((ext_vector_type(8)))  float  v8f;
typedef __attribute__((ext_vector_type(4)))  unsigned int u32x4;
typedef __attribute__((ext_vector_type(8)))  int i32x8;
typedef __attribute__((ext_vector_type(4)))  int i32x4;

#define EPSF 1e-5f

// ---------- dims ----------
// B=1, S=64, L=512, H=64, Dh=12, D=768, 3D=2304, DFF=3072, tokens T=32768
static constexpr int cS = 64, cL = 512, cH = 64, cDh = 12, cD = 768, cD3 = 2304, cDFF = 3072;
static constexpr int cT = cS * cL; // 32768

// ---------- helpers ----------
__device__ __forceinline__ u16b f2bf(float f) {
  unsigned u = __float_as_uint(f);
  unsigned r = 0x7FFFu + ((u >> 16) & 1u);
  return (u16b)((u + r) >> 16);
}

union FragBF { v16bf v; unsigned u[8]; };

// A-matrix fragment (16x32 bf16), per ISA 7.12.2:
// lanes 0-15: v0..3 -> K 0..7, v4..7 -> K 16..23 ; lanes 16-31: +8
__device__ __forceinline__ v16bf frag_a(const u16b* base, int ld, int lane) {
  int r  = lane & 15;
  int kh = (lane >> 4) << 3; // 0 or 8
  const u16b* p = base + r * ld + kh;
  FragBF f;
#pragma unroll
  for (int g = 0; g < 8; ++g) {
    int k = ((g & 4) << 2) + ((g & 3) << 1); // 0,2,4,6,16,18,20,22
    f.u[g] = *(const unsigned*)(p + k);
  }
  return f.v;
}

// B-matrix fragment (32x16 bf16), LDS stored transposed as [n][k]:
// lanes 0-15: v0..3 -> K 0..7, v4..7 -> K 16..23 ; lanes 16-31: +8
__device__ __forceinline__ v16bf frag_b(const u16b* base, int ld, int lane) {
  int n  = lane & 15;
  int kh = (lane >> 4) << 3; // 0 or 8
  const u16b* p = base + n * ld + kh;
  FragBF f;
#pragma unroll
  for (int g = 0; g < 8; ++g) f.u[g] = *(const unsigned*)(p + ((g & 4) << 2) + ((g & 3) << 1));
  return f.v;
}

__device__ __forceinline__ v8f wmma_bf16(v16bf a, v16bf b, v8f c) {
  return __builtin_amdgcn_wmma_f32_16x16x32_bf16(false, a, false, b, (short)0, c, false, false);
}

// ---------- Tensor Data Mover: load a 2-D bf16 tile (tile_k x tile_m) into LDS ----------
// Tile row = tile_k elements (64B = 16 DWORDs); pad_interval=3 (16 DW), pad_amount=0 (1 DW)
// => LDS row stride = 34 elements, matching the fragment loaders' LDS layout.
__device__ __forceinline__ void tdm_load_2d_bf16(unsigned lds_byte_addr,
                                                 unsigned long long gaddr,
                                                 unsigned tile_k, unsigned tile_m,
                                                 unsigned tensor_k, unsigned tensor_m,
                                                 unsigned stride_elems)
{
  u32x4 g0;
  g0.x = 1u;                                                     // count=1, user mode
  g0.y = lds_byte_addr;                                          // D# lds_addr
  g0.z = (unsigned)(gaddr & 0xFFFFFFFFull);                      // global_addr[31:0]
  g0.w = (unsigned)((gaddr >> 32) & 0x1FFFFFFull) | (2u << 30);  // [120:64] + type=2
  i32x8 g1;
  g1[0] = (int)((1u << 16) | (1u << 20) | (3u << 22) | (0u << 25)); // data_size=2B, pad_en, 16DW/1DW
  g1[1] = (int)((tensor_k & 0xFFFFu) << 16);                     // tensor_dim0[15:0]
  g1[2] = (int)((tensor_k >> 16) | ((tensor_m & 0xFFFFu) << 16)); // dim0[31:16] | dim1[15:0]
  g1[3] = (int)((tensor_m >> 16) | (tile_k << 16));              // dim1[31:16] | tile_dim0
  g1[4] = (int)tile_m;                                           // tile_dim1 (tile_dim2=0)
  g1[5] = (int)stride_elems;                                     // tensor_dim0_stride[31:0]
  g1[6] = 0;
  g1[7] = 0;
  i32x4 z4 = {0, 0, 0, 0};
#if defined(__clang_major__) && (__clang_major__ >= 23)
  i32x8 z8 = {0, 0, 0, 0, 0, 0, 0, 0};
  __builtin_amdgcn_tensor_load_to_lds(g0, g1, z4, z4, z8, 0);
#else
  __builtin_amdgcn_tensor_load_to_lds(g0, g1, z4, z4, 0);
#endif
}

// ---------- fp32 -> bf16 weight conversion ----------
__global__ __launch_bounds__(256) void cvt_bf16_kernel(const float* __restrict__ in,
                                                       u16b* __restrict__ out, int n) {
  int i = blockIdx.x * 256 + threadIdx.x;
  if (i < n) out[i] = f2bf(in[i]);
}

// ---------- generic tiled WMMA GEMM: C = A(MxK) * B(KxN) + bias ----------
// Block: 128 threads = 4 waves, tile 64(M) x 64(N), K step 32.
// A_BF16 path feeds the A tile via the Tensor Data Mover (async DMA + TENSORcnt).
template<bool A_BF16, bool RELU, bool WF32, bool WBF16>
__global__ __launch_bounds__(128) void gemm_wmma(
    const void* __restrict__ Ap, int lda,
    const u16b* __restrict__ B, int ldb,
    const float* __restrict__ bias,
    float* __restrict__ Cf, u16b* __restrict__ Cb, int ldc, int K)
{
  constexpr int LDS_A = 34, LDS_B = 34;
  __shared__ __align__(16) u16b As[64 * LDS_A];
  __shared__ __align__(16) u16b Bs[64 * LDS_B];
  const int tid = threadIdx.x, wave = tid >> 5, lane = tid & 31;
  const int tM = blockIdx.y << 6, tN = blockIdx.x << 6;
  const float* Af = (const float*)Ap;
  const u16b*  Ab = (const u16b*)Ap;

  v8f acc[4];
#pragma unroll
  for (int j = 0; j < 4; ++j)
#pragma unroll
    for (int r = 0; r < 8; ++r) acc[j][r] = 0.f;

  const int nk = K >> 5;
  for (int kb = 0; kb < nk; ++kb) {
    const int k0 = kb << 5;
    if constexpr (A_BF16) {
      // TDM: DMA the 64x32 bf16 A tile straight into LDS with row padding.
      if (wave == 0) {
        unsigned lds_a = (unsigned)(uintptr_t)(&As[0]);
        unsigned long long ga =
            (unsigned long long)(uintptr_t)(Ab + (size_t)tM * lda + k0);
        tdm_load_2d_bf16(lds_a, ga, 32u, 64u, (unsigned)K, (unsigned)cT, (unsigned)lda);
      }
    } else {
      // stage A 64x32 fp32 -> bf16
      int m = tid >> 1, kh = (tid & 1) << 4;
      size_t gro = (size_t)(tM + m) * lda + k0 + kh;
#pragma unroll
      for (int c = 0; c < 16; ++c) As[m * LDS_A + kh + c] = f2bf(Af[gro + c]);
    }
    { // stage B 32x64 into Bs[n][k]
      int k = tid >> 2, n0 = (tid & 3) << 4;
      size_t gro = (size_t)(k0 + k) * ldb + tN + n0;
#pragma unroll
      for (int u = 0; u < 16; ++u) Bs[(n0 + u) * LDS_B + k] = B[gro + u];
    }
    if (kb + 1 < nk) __builtin_prefetch((const void*)(B + (size_t)(k0 + 32) * ldb + tN), 0, 1);
    if constexpr (A_BF16) {
      if (wave == 0) __builtin_amdgcn_s_wait_tensorcnt(0);
    }
    __syncthreads();
    v16bf a = frag_a(As + (wave << 4) * LDS_A, LDS_A, lane);
#pragma unroll
    for (int j = 0; j < 4; ++j) {
      v16bf b = frag_b(Bs + (j << 4) * LDS_B, LDS_B, lane);
      acc[j] = wmma_bf16(a, b, acc[j]);
    }
    __syncthreads();
  }

  const int r0 = tM + (wave << 4) + ((lane >> 4) << 3);
  const int cl = lane & 15;
#pragma unroll
  for (int j = 0; j < 4; ++j) {
    int c = tN + (j << 4) + cl;
    float bv = bias ? bias[c] : 0.f;
#pragma unroll
    for (int r = 0; r < 8; ++r) {
      float v = acc[j][r] + bv;
      if (RELU) v = fmaxf(v, 0.f);
      size_t off = (size_t)(r0 + r) * ldc + c;
      if (WF32)  Cf[off] = v;
      if (WBF16) Cb[off] = f2bf(v);
    }
  }
}

// ---------- row attention scores: S[h,i,j] = sum_{s,c} Q[s,i,h,c] K[s,j,h,c] ----------
// grid: (jTile=8, iTile=8, h=64); K contraction = S*Dh = 768 (strided gather, no TDM)
__global__ __launch_bounds__(128) void row_scores_kernel(const u16b* __restrict__ qkv,
                                                         float* __restrict__ sc)
{
  constexpr int LDS_A = 34, LDS_B = 34;
  __shared__ __align__(16) u16b As[64 * LDS_A];
  __shared__ __align__(16) u16b Bs[64 * LDS_B];
  const int tid = threadIdx.x, wave = tid >> 5, lane = tid & 31;
  const int tI = blockIdx.y << 6, tJ = blockIdx.x << 6, h = blockIdx.z;

  v8f acc[4];
#pragma unroll
  for (int j = 0; j < 4; ++j)
#pragma unroll
    for (int r = 0; r < 8; ++r) acc[j][r] = 0.f;

  for (int kb = 0; kb < 24; ++kb) {
    const int k0 = kb << 5;
    { // A = Q gather over t=(s,c)
      int m = tid >> 1, kh = (tid & 1) << 4;
#pragma unroll
      for (int c = 0; c < 16; ++c) {
        int kk = k0 + kh + c, s = kk / 12, cc = kk - 12 * s;
        As[m * LDS_A + kh + c] =
            qkv[(size_t)(s * cL + tI + m) * cD3 + h * cDh + cc];
      }
    }
    { // B = K gather, stored Bs[j][t]
      int k = tid >> 2, n0 = (tid & 3) << 4;
      int kk = k0 + k, s = kk / 12, cc = kk - 12 * s;
#pragma unroll
      for (int u = 0; u < 16; ++u)
        Bs[(n0 + u) * LDS_B + k] =
            qkv[(size_t)(s * cL + tJ + n0 + u) * cD3 + cD + h * cDh + cc];
    }
    __syncthreads();
    v16bf a = frag_a(As + (wave << 4) * LDS_A, LDS_A, lane);
#pragma unroll
    for (int j = 0; j < 4; ++j) {
      v16bf b = frag_b(Bs + (j << 4) * LDS_B, LDS_B, lane);
      acc[j] = wmma_bf16(a, b, acc[j]);
    }
    __syncthreads();
  }
  const int r0 = tI + (wave << 4) + ((lane >> 4) << 3);
  const int cl = lane & 15;
#pragma unroll
  for (int j = 0; j < 4; ++j)
#pragma unroll
    for (int r = 0; r < 8; ++r)
      sc[((size_t)h * cL + r0 + r) * cL + tJ + (j << 4) + cl] = acc[j][r];
}

// ---------- row softmax over j (length 512), fp32 in -> bf16 out ----------
__global__ __launch_bounds__(256) void softmax_row_kernel(const float* __restrict__ sc,
                                                          u16b* __restrict__ at)
{
  __shared__ float red[256];
  const size_t row = (size_t)blockIdx.x * cL;
  const int t = threadIdx.x;
  float a = sc[row + t], b = sc[row + t + 256];
  red[t] = fmaxf(a, b);
  __syncthreads();
  for (int s2 = 128; s2 > 0; s2 >>= 1) { if (t < s2) red[t] = fmaxf(red[t], red[t + s2]); __syncthreads(); }
  float m = red[0];
  __syncthreads();
  float e0 = __expf(a - m), e1 = __expf(b - m);
  red[t] = e0 + e1;
  __syncthreads();
  for (int s2 = 128; s2 > 0; s2 >>= 1) { if (t < s2) red[t] += red[t + s2]; __syncthreads(); }
  float inv = 1.f / red[0];
  at[row + t]       = f2bf(e0 * inv);
  at[row + t + 256] = f2bf(e1 * inv);
}

// ---------- row AV: out[s,i,h,d] = sum_j attn[h,i,j] V[s,j,h,d] ----------
// grid: (iTile=8, s=64, h=64); K contraction = L = 512; N=16 (d padded from 12)
// A tile (attention probs) is contiguous bf16 -> fed by the Tensor Data Mover.
__global__ __launch_bounds__(128) void row_av_kernel(const u16b* __restrict__ at,
                                                     const u16b* __restrict__ qkv,
                                                     float* __restrict__ ro)
{
  constexpr int LDS_A = 34, LDS_B = 34;
  __shared__ __align__(16) u16b As[64 * LDS_A];
  __shared__ __align__(16) u16b Bs[16 * LDS_B];
  const int tid = threadIdx.x, wave = tid >> 5, lane = tid & 31;
  const int tI = blockIdx.x << 6, s = blockIdx.y, h = blockIdx.z;

  v8f acc;
#pragma unroll
  for (int r = 0; r < 8; ++r) acc[r] = 0.f;

  for (int kb = 0; kb < 16; ++kb) {
    const int j0 = kb << 5;
    // A = attn tile via TDM (rows contiguous in j, stride L)
    if (wave == 0) {
      unsigned lds_a = (unsigned)(uintptr_t)(&As[0]);
      unsigned long long ga =
          (unsigned long long)(uintptr_t)(at + (size_t)(h * cL + tI) * cL + j0);
      tdm_load_2d_bf16(lds_a, ga, 32u, 64u, (unsigned)cL, (unsigned)cT, (unsigned)cL);
    }
    { // B = V tile, Bs[d][j], d padded to 16
      int d = tid >> 3, j4 = (tid & 7) << 2;
#pragma unroll
      for (int u = 0; u < 4; ++u) {
        int jj = j0 + j4 + u;
        Bs[d * LDS_B + j4 + u] =
            (d < cDh) ? qkv[(size_t)(s * cL + jj) * cD3 + 2 * cD + h * cDh + d] : (u16b)0;
      }
    }
    if (wave == 0) __builtin_amdgcn_s_wait_tensorcnt(0);
    __syncthreads();
    v16bf a = frag_a(As + (wave << 4) * LDS_A, LDS_A, lane);
    v16bf b = frag_b(Bs, LDS_B, lane);
    acc = wmma_bf16(a, b, acc);
    __syncthreads();
  }
  const int i0 = tI + (wave << 4) + ((lane >> 4) << 3);
  const int d = lane & 15;
  if (d < cDh) {
#pragma unroll
    for (int r = 0; r < 8; ++r)
      ro[(size_t)(s * cL + i0 + r) * cD + h * cDh + d] = acc[r];
  }
}

// ---------- add + LayerNorm over D=768: out = LN(a+b)*g+be ----------
__global__ __launch_bounds__(256) void add_ln_kernel(const float* __restrict__ A,
                                                     const float* __restrict__ Bv,
                                                     const float* __restrict__ g,
                                                     const float* __restrict__ be,
                                                     float* __restrict__ out)
{
  __shared__ float red[256];
  const size_t base = (size_t)blockIdx.x * cD;
  const int t = threadIdx.x;
  float v[3], sum = 0.f;
#pragma unroll
  for (int k = 0; k < 3; ++k) {
    int idx = t + (k << 8);
    v[k] = A[base + idx] + Bv[base + idx];
    sum += v[k];
  }
  red[t] = sum;
  __syncthreads();
  for (int s2 = 128; s2 > 0; s2 >>= 1) { if (t < s2) red[t] += red[t + s2]; __syncthreads(); }
  const float mean = red[0] * (1.f / cD);
  __syncthreads();
  float sq = 0.f;
#pragma unroll
  for (int k = 0; k < 3; ++k) { float d = v[k] - mean; sq += d * d; }
  red[t] = sq;
  __syncthreads();
  for (int s2 = 128; s2 > 0; s2 >>= 1) { if (t < s2) red[t] += red[t + s2]; __syncthreads(); }
  const float rstd = rsqrtf(red[0] * (1.f / cD) + EPSF);
#pragma unroll
  for (int k = 0; k < 3; ++k) {
    int idx = t + (k << 8);
    out[base + idx] = (v[k] - mean) * rstd * g[idx] + be[idx];
  }
}

// ---------- fused column attention: per (l,h): softmax_j(Q K^T) V, all 64x64 in LDS ----------
// grid: (l=512, h=64); 128 threads = 4 waves
__global__ __launch_bounds__(128) void col_attn_kernel(const u16b* __restrict__ qkv,
                                                       float* __restrict__ co)
{
  __shared__ __align__(16) u16b Qs[64 * 34];
  __shared__ __align__(16) u16b Ks[64 * 34];
  __shared__ __align__(16) u16b Vs[16 * 66];
  __shared__ __align__(16) u16b Ps[64 * 66];
  __shared__ float Sf[64 * 65];
  const int tid = threadIdx.x, wave = tid >> 5, lane = tid & 31;
  const int l = blockIdx.x, h = blockIdx.y;

  { // stage Q,K: 64 rows x 32 (c padded from 12)
    int m = tid >> 1, kh = (tid & 1) << 4;
    size_t qo = (size_t)(m * cL + l) * cD3 + h * cDh;
#pragma unroll
    for (int c = 0; c < 16; ++c) {
      int cc = kh + c;
      Qs[m * 34 + cc] = (cc < cDh) ? qkv[qo + cc] : (u16b)0;
      Ks[m * 34 + cc] = (cc < cDh) ? qkv[qo + cD + cc] : (u16b)0;
    }
  }
  { // stage V: Vs[d][j], 16x64
    int d = tid >> 3, j8 = (tid & 7) << 3;
#pragma unroll
    for (int u = 0; u < 8; ++u) {
      int j = j8 + u;
      Vs[d * 66 + j] =
          (d < cDh) ? qkv[(size_t)(j * cL + l) * cD3 + 2 * cD + h * cDh + d] : (u16b)0;
    }
  }
  __syncthreads();

  // scores 64x64 (single K-step of 32, c>=12 zero-padded)
  v8f acc[4];
#pragma unroll
  for (int j = 0; j < 4; ++j)
#pragma unroll
    for (int r = 0; r < 8; ++r) acc[j][r] = 0.f;
  v16bf a = frag_a(Qs + (wave << 4) * 34, 34, lane);
#pragma unroll
  for (int jb = 0; jb < 4; ++jb) {
    v16bf b = frag_b(Ks + (jb << 4) * 34, 34, lane);
    acc[jb] = wmma_bf16(a, b, acc[jb]);
  }
  const int i0 = (wave << 4) + ((lane >> 4) << 3);
  const int cl = lane & 15;
#pragma unroll
  for (int jb = 0; jb < 4; ++jb)
#pragma unroll
    for (int r = 0; r < 8; ++r) Sf[(i0 + r) * 65 + (jb << 4) + cl] = acc[jb][r];
  __syncthreads();

  if (tid < 64) { // softmax over j for row i=tid
    float mx = -1e30f;
    for (int j = 0; j < 64; ++j) mx = fmaxf(mx, Sf[tid * 65 + j]);
    float sm = 0.f;
    for (int j = 0; j < 64; ++j) sm += __expf(Sf[tid * 65 + j] - mx);
    float inv = 1.f / sm;
    for (int j = 0; j < 64; ++j) Ps[tid * 66 + j] = f2bf(__expf(Sf[tid * 65 + j] - mx) * inv);
  }
  __syncthreads();

  // O = P (64x64) @ V (64x16), 2 K-steps of 32
  v8f o;
#pragma unroll
  for (int r = 0; r < 8; ++r) o[r] = 0.f;
  for (int j0 = 0; j0 < 64; j0 += 32) {
    v16bf a2 = frag_a(Ps + (wave << 4) * 66 + j0, 66, lane);
    v16bf b2 = frag_b(Vs + j0, 66, lane);
    o = wmma_bf16(a2, b2, o);
  }
  const int d = lane & 15;
  if (d < cDh) {
#pragma unroll
    for (int r = 0; r < 8; ++r)
      co[(size_t)((i0 + r) * cL + l) * cD + h * cDh + d] = o[r];
  }
}

// ---------- workspace layout (bytes) ----------
static constexpr size_t OFF_WROW  = 0;                               // 768*2304*2
static constexpr size_t OFF_WCOL  = OFF_WROW + (size_t)cD * cD3 * 2;
static constexpr size_t OFF_W1    = OFF_WCOL + (size_t)cD * cD3 * 2;
static constexpr size_t OFF_W2    = OFF_W1 + (size_t)cD * cDFF * 2;
static constexpr size_t OFF_QKV   = OFF_W2 + (size_t)cDFF * cD * 2;  // 32768*2304 bf16
static constexpr size_t OFF_SCORE = OFF_QKV + (size_t)cT * cD3 * 2;  // 64*512*512 f32
static constexpr size_t OFF_ATTN  = OFF_SCORE + (size_t)cH * cL * cL * 4; // bf16
static constexpr size_t OFF_BUFA  = OFF_ATTN + (size_t)cH * cL * cL * 2;  // f32 T*D (rowout/colout/ff2)
static constexpr size_t OFF_OUT1  = OFF_BUFA + (size_t)cT * cD * 4;
static constexpr size_t OFF_OUT2  = OFF_OUT1 + (size_t)cT * cD * 4;
static constexpr size_t OFF_Y     = OFF_OUT2 + (size_t)cT * cD * 4;
// FFN hidden (32768*3072 bf16 = 192MB) overlays dead QKV+SCORE region (218MB)
static constexpr size_t OFF_HBF   = OFF_QKV;

extern "C" void kernel_launch(void* const* d_in, const int* in_sizes, int n_in,
                              void* d_out, int out_size, void* d_ws, size_t ws_size,
                              hipStream_t stream)
{
  const float* x     = (const float*)d_in[0];
  const float* w_row = (const float*)d_in[1];
  const float* b_row = (const float*)d_in[2];
  const float* w_col = (const float*)d_in[3];
  const float* b_col = (const float*)d_in[4];
  const float* g_a1  = (const float*)d_in[5];
  const float* be_a1 = (const float*)d_in[6];
  const float* g_a2  = (const float*)d_in[7];
  const float* be_a2 = (const float*)d_in[8];
  const float* w1    = (const float*)d_in[9];
  const float* b1f   = (const float*)d_in[10];
  const float* w2    = (const float*)d_in[11];
  const float* b2f   = (const float*)d_in[12];
  const float* g_n1  = (const float*)d_in[13];
  const float* be_n1 = (const float*)d_in[14];
  const float* g_n2  = (const float*)d_in[15];
  const float* be_n2 = (const float*)d_in[16];

  char* ws = (char*)d_ws;
  u16b* wrow_bf = (u16b*)(ws + OFF_WROW);
  u16b* wcol_bf = (u16b*)(ws + OFF_WCOL);
  u16b* w1_bf   = (u16b*)(ws + OFF_W1);
  u16b* w2_bf   = (u16b*)(ws + OFF_W2);
  u16b* qkv     = (u16b*)(ws + OFF_QKV);
  float* score  = (float*)(ws + OFF_SCORE);
  u16b* attn    = (u16b*)(ws + OFF_ATTN);
  float* bufA   = (float*)(ws + OFF_BUFA);
  float* out1   = (float*)(ws + OFF_OUT1);
  float* out2   = (float*)(ws + OFF_OUT2);
  float* ybuf   = (float*)(ws + OFF_Y);
  u16b* hbf     = (u16b*)(ws + OFF_HBF);
  float* outp   = (float*)d_out;

  auto cvt = [&](const float* src, u16b* dst, int n) {
    cvt_bf16_kernel<<<(n + 255) / 256, 256, 0, stream>>>(src, dst, n);
  };
  cvt(w_row, wrow_bf, cD * cD3);
  cvt(w_col, wcol_bf, cD * cD3);
  cvt(w1,    w1_bf,   cD * cDFF);
  cvt(w2,    w2_bf,   cDFF * cD);

  // row QKV: qkv = bf16(x @ w_row + b_row), [32768 x 2304]
  gemm_wmma<false, false, false, true><<<dim3(cD3 / 64, cT / 64), 128, 0, stream>>>(
      (const void*)x, cD, wrow_bf, cD3, b_row, nullptr, qkv, cD3, cD);

  // row attention
  row_scores_kernel<<<dim3(8, 8, cH), 128, 0, stream>>>(qkv, score);
  softmax_row_kernel<<<cH * cL, 256, 0, stream>>>(score, attn);
  row_av_kernel<<<dim3(8, cS, cH), 128, 0, stream>>>(attn, qkv, bufA);
  add_ln_kernel<<<cT, 256, 0, stream>>>(x, bufA, g_a1, be_a1, out1);

  // col QKV: qkv = bf16(out1 @ w_col + b_col)
  gemm_wmma<false, false, false, true><<<dim3(cD3 / 64, cT / 64), 128, 0, stream>>>(
      (const void*)out1, cD, wcol_bf, cD3, b_col, nullptr, qkv, cD3, cD);

  // fused col attention
  col_attn_kernel<<<dim3(cL, cH), 128, 0, stream>>>(qkv, bufA);
  add_ln_kernel<<<cT, 256, 0, stream>>>(out1, bufA, g_a2, be_a2, out2);

  // y = LN(x + out2)
  add_ln_kernel<<<cT, 256, 0, stream>>>(x, out2, g_n1, be_n1, ybuf);

  // FFN: hbf = bf16(relu(y @ w1 + b1f)) ; ff2 = hbf @ w2 + b2f (A tile via TDM)
  gemm_wmma<false, true, false, true><<<dim3(cDFF / 64, cT / 64), 128, 0, stream>>>(
      (const void*)ybuf, cD, w1_bf, cDFF, b1f, nullptr, hbf, cDFF, cD);
  gemm_wmma<true, false, true, false><<<dim3(cD / 64, cT / 64), 128, 0, stream>>>(
      (const void*)hbf, cDFF, w2_bf, cD, b2f, bufA, nullptr, cD, cDFF);

  // final: out = LN(y + ff2)
  add_ln_kernel<<<cT, 256, 0, stream>>>(ybuf, bufA, g_n2, be_n2, outp);

  (void)in_sizes; (void)n_in; (void)out_size; (void)ws_size;
}
